// CausalSelfAttention_82068235091988
// MI455X (gfx1250) — compile-verified, matrix operand order checked by reading, not run on hardware
//
#include <hip/hip_runtime.h>

typedef __attribute__((ext_vector_type(16))) __bf16 v16bf;
typedef __attribute__((ext_vector_type(8)))  float  v8f;
typedef __attribute__((ext_vector_type(4)))  unsigned u32x4;
typedef __attribute__((ext_vector_type(8)))  unsigned u32x8;

// ---------- bf16 helpers (manual RNE conversion) ---------------------------
__device__ inline __bf16 f2bf(float f) {
    union { float f; unsigned u; } a; a.f = f;
    unsigned r = a.u + 0x7FFFu + ((a.u >> 16) & 1u);
    unsigned short h = (unsigned short)(r >> 16);
    union { unsigned short s; __bf16 b; } c; c.s = h;
    return c.b;
}
__device__ inline float bf2f(__bf16 b) {
    union { unsigned short s; __bf16 b; } c; c.b = b;
    union { unsigned u; float f; } a; a.u = ((unsigned)c.s) << 16;
    return a.f;
}

union FragU { v16bf v; uint4 q[2]; };
__device__ inline v16bf load_frag(const __bf16* lo, const __bf16* hi) {
    FragU f;
    f.q[0] = *(const uint4*)lo;   // fragment elems 0..7   (K pattern low half)
    f.q[1] = *(const uint4*)hi;   // fragment elems 8..15  (K pattern high half)
    return f.v;
}
union B16x8 { uint4 q; __bf16 h[8]; };

// ---------------------------------------------------------------------------
// Kernel 1: fp32 -> bf16 elementwise convert
// ---------------------------------------------------------------------------
__global__ void cvt_f32_bf16(const float* __restrict__ s, __bf16* __restrict__ d, long n) {
    long i = (long)blockIdx.x * blockDim.x + threadIdx.x;
    long stride = (long)gridDim.x * blockDim.x;
    for (; i < n; i += stride) d[i] = f2bf(s[i]);
}

// ---------------------------------------------------------------------------
// Kernel 2: tiled bf16 GEMM, fp32 accumulate. C[M,N] = A[M,K] * B[K,N].
// BM=128 BN=128 BK=32; 256 threads = 8 waves in a 4(M) x 2(N) grid,
// each wave computes a 32x64 tile = 2x4 WMMA fragments.
// Double-buffered LDS with register prefetch: one barrier per K iteration,
// global loads for tile i+1 overlap WMMAs on tile i.
// ---------------------------------------------------------------------------
template<bool STORE_BF16>
__global__ __launch_bounds__(256) void gemm_bf16(const __bf16* __restrict__ A,
                                                 const __bf16* __restrict__ B,
                                                 void* __restrict__ Cp,
                                                 int M, int N, int K) {
    constexpr int BM = 128, BN = 128, BK = 32, LDP = BK + 8; // 80B stride, conflict-free
    __shared__ __bf16 As[2][BM][LDP];   // (m, k)
    __shared__ __bf16 Bs[2][BN][LDP];   // (n, k)  -- B stored transposed

    const int bm = blockIdx.y * BM;
    const int bn = blockIdx.x * BN;
    const int tid  = threadIdx.x;
    const int wave = tid >> 5;
    const int lane = tid & 31;
    const int half = lane >> 4;
    const int ln   = lane & 15;
    const int wm = wave & 3;   // 0..3 -> 32 rows each
    const int wn = wave >> 2;  // 0..1 -> 64 cols each

    const v8f vzero = {0.f,0.f,0.f,0.f,0.f,0.f,0.f,0.f};
    v8f acc[2][4];
    #pragma unroll
    for (int im = 0; im < 2; ++im)
        #pragma unroll
        for (int ni = 0; ni < 4; ++ni) acc[im][ni] = vzero;

    const int ar = tid >> 1, ac = (tid & 1) * 16;       // A tile: 16 elems/thread
    const int bk = tid & 31, bn0 = (tid >> 5) * 16;     // B tile: 16 elems/thread
    const __bf16* Arow  = A + (long)(bm + ar) * K + ac;
    const __bf16* Bbase = B + bn + bn0;

    uint4 a0, a1; B16x8 b0, b1;
    { // preload tile 0 into registers
        const uint4* g  = (const uint4*)Arow;
        a0 = g[0]; a1 = g[1];
        const uint4* gb = (const uint4*)(Bbase + (long)bk * N);
        b0.q = gb[0]; b1.q = gb[1];
    }
    { // stage tile 0 into buffer 0
        *(uint4*)&As[0][ar][ac]     = a0;
        *(uint4*)&As[0][ar][ac + 8] = a1;
        #pragma unroll
        for (int j = 0; j < 8; ++j) Bs[0][bn0 + j][bk]     = b0.h[j];
        #pragma unroll
        for (int j = 0; j < 8; ++j) Bs[0][bn0 + 8 + j][bk] = b1.h[j];
    }
    __syncthreads();

    int buf = 0;
    for (int k0 = 0; k0 < K; k0 += BK) {
        const bool more = (k0 + BK) < K;
        if (more) { // global prefetch of next tile overlaps this tile's WMMAs
            const uint4* g  = (const uint4*)(Arow + k0 + BK);
            a0 = g[0]; a1 = g[1];
            const uint4* gb = (const uint4*)(Bbase + (long)(k0 + BK + bk) * N);
            b0.q = gb[0]; b1.q = gb[1];
        }

        v16bf af[2], bf_[4];
        #pragma unroll
        for (int im = 0; im < 2; ++im) {
            const __bf16* r = &As[buf][wm * 32 + im * 16 + ln][0];
            af[im] = load_frag(r + half * 8, r + 16 + half * 8);
        }
        #pragma unroll
        for (int ni = 0; ni < 4; ++ni) {
            const __bf16* r = &Bs[buf][wn * 64 + ni * 16 + ln][0];
            bf_[ni] = load_frag(r + half * 8, r + 16 + half * 8);
        }
        #pragma unroll
        for (int im = 0; im < 2; ++im)
            #pragma unroll
            for (int ni = 0; ni < 4; ++ni)
                acc[im][ni] = __builtin_amdgcn_wmma_f32_16x16x32_bf16(
                    false, af[im], false, bf_[ni], (short)0, acc[im][ni], false, false);

        if (more) { // stage next tile into the other buffer, single barrier
            *(uint4*)&As[buf ^ 1][ar][ac]     = a0;
            *(uint4*)&As[buf ^ 1][ar][ac + 8] = a1;
            #pragma unroll
            for (int j = 0; j < 8; ++j) Bs[buf ^ 1][bn0 + j][bk]     = b0.h[j];
            #pragma unroll
            for (int j = 0; j < 8; ++j) Bs[buf ^ 1][bn0 + 8 + j][bk] = b1.h[j];
            __syncthreads();
            buf ^= 1;
        }
    }

    #pragma unroll
    for (int im = 0; im < 2; ++im)
        #pragma unroll
        for (int ni = 0; ni < 4; ++ni)
            #pragma unroll
            for (int i = 0; i < 8; ++i) {
                int m = bm + wm * 32 + im * 16 + 8 * half + i;
                int n = bn + wn * 64 + ni * 16 + ln;
                if (STORE_BF16) ((__bf16*)Cp)[(long)m * N + n] = f2bf(acc[im][ni][i]);
                else            ((float*)Cp)[(long)m * N + n] = acc[im][ni][i];
            }
}

// ---------------------------------------------------------------------------
// Kernel 3: RoPE on q,k + reshape qkv(B,T,3C) -> q/k/v (B,H,T,hd), bf16.
// ---------------------------------------------------------------------------
__global__ __launch_bounds__(256) void rope_reshape(const __bf16* __restrict__ qkv,
                                                    const float* __restrict__ fr,
                                                    __bf16* __restrict__ Qo,
                                                    __bf16* __restrict__ Ko,
                                                    __bf16* __restrict__ Vo) {
    int i  = blockIdx.x * 256 + threadIdx.x;     // 2*2048*16*64 threads
    int d2 = i & 63;
    int h  = (i >> 6) & 15;
    int t  = (i >> 10) & 2047;
    int b  = i >> 21;
    const __bf16* row = qkv + ((long)(b * 2048 + t)) * 6144 + h * 128;
    float c = fr[((long)t * 64 + d2) * 2 + 0];
    float s = fr[((long)t * 64 + d2) * 2 + 1];
    long  o = (((long)(b * 16 + h)) * 2048 + t) * 128 + 2 * d2;
    {
        float re = bf2f(row[2 * d2]), im = bf2f(row[2 * d2 + 1]);
        Qo[o] = f2bf(re * c - im * s); Qo[o + 1] = f2bf(re * s + im * c);
    }
    {
        float re = bf2f(row[2048 + 2 * d2]), im = bf2f(row[2048 + 2 * d2 + 1]);
        Ko[o] = f2bf(re * c - im * s); Ko[o + 1] = f2bf(re * s + im * c);
    }
    Vo[o] = row[4096 + 2 * d2]; Vo[o + 1] = row[4096 + 2 * d2 + 1];
}

// ---------------------------------------------------------------------------
// Kernel 4: flash attention with prefix-LM mask.
// 128 threads = 4 waves; each wave = 16 q rows; WG = 64 q rows of one (b,h).
// K block (32x128 bf16) staged by the Tensor Data Mover into padded LDS;
// V staged manually (needs transpose). Online softmax in fp32.
// ---------------------------------------------------------------------------
__global__ __launch_bounds__(128) void attn_kernel(const __bf16* __restrict__ Q,
                                                   const __bf16* __restrict__ K,
                                                   const __bf16* __restrict__ V,
                                                   __bf16* __restrict__ Y) {
    constexpr int T = 2048, HD = 128, KB = 32;
    const int numQT = T / 64;
    const int qt = blockIdx.x % numQT;
    const int bh = blockIdx.x / numQT;           // 0..31  (= b*16 + h)
    const int qb = qt * 64;
    const int tid  = threadIdx.x;
    const int wave = tid >> 5;
    const int lane = tid & 31;
    const int half = lane >> 4;
    const int ln   = lane & 15;

    const __bf16* Qh = Q + (long)bh * T * HD;
    const __bf16* Kh = K + (long)bh * T * HD;
    const __bf16* Vh = V + (long)bh * T * HD;

    __shared__ __bf16 Ks[KB][HD + 8];     // (key, d), 272B row stride (TDM-padded)
    __shared__ __bf16 Vs[HD][KB + 8];     // (d, key) transposed, 80B row stride
    __shared__ __bf16 Ps[4][16][KB + 8];  // per-wave P tile re-layout

    // Q A-fragments for hd = 4 chunks of K=32, loaded once from global.
    v16bf qf[4];
    {
        const __bf16* qrow = Qh + (long)(qb + wave * 16 + ln) * HD;
        #pragma unroll
        for (int c = 0; c < 4; ++c)
            qf[c] = load_frag(qrow + c * 32 + half * 8, qrow + c * 32 + 16 + half * 8);
    }

    const v8f vzero = {0.f,0.f,0.f,0.f,0.f,0.f,0.f,0.f};
    v8f o[8];
    #pragma unroll
    for (int f = 0; f < 8; ++f) o[f] = vzero;
    float mstat[8], lstat[8];
    #pragma unroll
    for (int i = 0; i < 8; ++i) { mstat[i] = -1e30f; lstat[i] = 0.f; }

    const float scale = 0.08838834764831845f;  // 1/sqrt(128)
    const int kend = (qb + 64 <= 1024) ? 1024 : (qb + 64);
    const unsigned ks_lds = (unsigned)(unsigned long)(void*)&Ks[0][0];

    for (int kbase = 0; kbase < kend; kbase += KB) {
        __syncthreads();
        if (wave == 0) {
            // ---- Tensor Data Mover: 2D tile 128(d) x 32(key), 2B elements,
            // pad 4 DWORDs after every 64 DWORDs (row) -> LDS stride 272B.
            unsigned long ga = (unsigned long)(const void*)(Kh + (long)kbase * HD);
            u32x4 g0;
            g0.x = 1u;                                  // count=1, user descriptor
            g0.y = ks_lds;                              // lds_addr
            g0.z = (unsigned)ga;                        // global_addr[31:0]
            g0.w = ((unsigned)(ga >> 32) & 0x1FFFFFFu)  // global_addr[56:32]
                   | (2u << 30);                        // type=2 ("image")
            u32x8 g1;
            g1.s0 = (1u << 16)        // data_size = 2 bytes
                  | (1u << 20)        // pad_enable
                  | (5u << 22)        // pad_interval: 64 DWORDs (one 256B row)
                  | (3u << 25);       // pad_amount: 4 DWORDs (16B)
            g1.s1 = (128u << 16);     // tensor_dim0 = 128 (bits 79:48)
            g1.s2 = (2048u << 16);    // tensor_dim1 = 2048 (bits 111:80)
            g1.s3 = (128u << 16);     // tile_dim0 = 128 (bits 127:112)
            g1.s4 = 32u;              // tile_dim1 = 32, tile_dim2 = 0
            g1.s5 = 128u;             // tensor_dim0_stride = 128 elements
            g1.s6 = 0u;
            g1.s7 = 0u;
            asm volatile("tensor_load_to_lds %0, %1" :: "s"(g0), "s"(g1) : "memory");
            __builtin_amdgcn_s_wait_tensorcnt(0);
        }
        { // stage V block transposed: (d, key)
            int key = tid & 31, dg = tid >> 5;   // dg in 0..3
            const __bf16* g = Vh + (long)(kbase + key) * HD + dg * 32;
            #pragma unroll
            for (int j = 0; j < 32; ++j) Vs[dg * 32 + j][key] = g[j];
        }
        __syncthreads();

        // ---- S = Q * K^T  (16 x 32, two N=16 fragments, 4 K-chunk accums)
        v8f s[2]; s[0] = vzero; s[1] = vzero;
        #pragma unroll
        for (int c = 0; c < 4; ++c)
            #pragma unroll
            for (int f = 0; f < 2; ++f) {
                const __bf16* kr = &Ks[f * 16 + ln][0];
                v16bf kb = load_frag(kr + c * 32 + half * 8, kr + c * 32 + 16 + half * 8);
                s[f] = __builtin_amdgcn_wmma_f32_16x16x32_bf16(
                    false, qf[c], false, kb, (short)0, s[f], false, false);
            }

        // ---- mask + online softmax (rows: vgpr i -> m = 8*half + i)
        float xv[2][8];
        #pragma unroll
        for (int f = 0; f < 2; ++f)
            #pragma unroll
            for (int i = 0; i < 8; ++i) {
                int r  = qb + wave * 16 + 8 * half + i;
                int kk = kbase + f * 16 + ln;
                bool ok = (kk <= r) || ((r < 1024) && (kk < 1024));
                xv[f][i] = ok ? s[f][i] * scale : -1e30f;
            }
        float alph[8], pv0[8], pv1[8];
        #pragma unroll
        for (int i = 0; i < 8; ++i) {
            float mx = fmaxf(xv[0][i], xv[1][i]);
            #pragma unroll
            for (int m = 1; m < 16; m <<= 1) mx = fmaxf(mx, __shfl_xor(mx, m, 32));
            float mnew = fmaxf(mstat[i], mx);
            alph[i] = __expf(mstat[i] - mnew);
            float p0 = __expf(xv[0][i] - mnew);
            float p1 = __expf(xv[1][i] - mnew);
            float sum = p0 + p1;
            #pragma unroll
            for (int m = 1; m < 16; m <<= 1) sum += __shfl_xor(sum, m, 32);
            lstat[i] = lstat[i] * alph[i] + sum;
            mstat[i] = mnew;
            pv0[i] = p0; pv1[i] = p1;
        }
        v8f av;
        #pragma unroll
        for (int i = 0; i < 8; ++i) av[i] = alph[i];
        #pragma unroll
        for (int f = 0; f < 8; ++f) o[f] *= av;

        // ---- re-layout P (D layout -> A layout) through per-wave LDS
        #pragma unroll
        for (int i = 0; i < 8; ++i) {
            Ps[wave][8 * half + i][0 * 16 + ln] = f2bf(pv0[i]);
            Ps[wave][8 * half + i][1 * 16 + ln] = f2bf(pv1[i]);
        }
        const __bf16* pr = &Ps[wave][ln][0];
        v16bf pa = load_frag(pr + half * 8, pr + 16 + half * 8);

        // ---- O += P * V   (8 hd-chunks of N=16)
        #pragma unroll
        for (int f = 0; f < 8; ++f) {
            const __bf16* vr = &Vs[f * 16 + ln][0];
            v16bf vb = load_frag(vr + half * 8, vr + 16 + half * 8);
            o[f] = __builtin_amdgcn_wmma_f32_16x16x32_bf16(
                false, pa, false, vb, (short)0, o[f], false, false);
        }
    }

    // ---- normalize and write y (B,T,C) as bf16 for the final GEMM
    float inv[8];
    #pragma unroll
    for (int i = 0; i < 8; ++i) inv[i] = 1.f / lstat[i];
    const int b = bh >> 4, h = bh & 15;
    #pragma unroll
    for (int f = 0; f < 8; ++f)
        #pragma unroll
        for (int i = 0; i < 8; ++i) {
            int t   = qb + wave * 16 + 8 * half + i;
            int col = h * 128 + f * 16 + ln;
            Y[((long)(b * 2048 + t)) * 2048 + col] = f2bf(o[f][i] * inv[i]);
        }
}

// ---------------------------------------------------------------------------
// Launch
// ---------------------------------------------------------------------------
extern "C" void kernel_launch(void* const* d_in, const int* in_sizes, int n_in,
                              void* d_out, int out_size, void* d_ws, size_t ws_size,
                              hipStream_t stream) {
    (void)in_sizes; (void)n_in; (void)out_size; (void)ws_size;
    const float* x      = (const float*)d_in[0];  // (2,2048,2048)
    const float* w_qkv  = (const float*)d_in[1];  // (2048,6144)
    const float* w_proj = (const float*)d_in[2];  // (2048,2048)
    const float* freqs  = (const float*)d_in[3];  // (2048,64,2)
    // d_in[4] = attn_mask: prefix mask computed analytically in-kernel.
    float* out = (float*)d_out;

    char* ws = (char*)d_ws;
    size_t off = 0;
    auto take = [&](long elems) -> __bf16* {
        __bf16* p = (__bf16*)(ws + off);
        off += ((size_t)elems * 2 + 255) & ~(size_t)255;
        return p;
    };
    __bf16* xb     = take(4096L * 2048);   // x bf16
    __bf16* wqkvb  = take(2048L * 6144);
    __bf16* wprojb = take(2048L * 2048);
    __bf16* qkvb   = take(4096L * 6144);   // qkv bf16 (B,T,3C)
    __bf16* qbuf   = take(32L * 2048 * 128);
    __bf16* kbuf   = take(32L * 2048 * 128);
    __bf16* vbuf   = take(32L * 2048 * 128);
    __bf16* yb     = take(4096L * 2048);   // attention output (B,T,C)

    cvt_f32_bf16<<<2048, 256, 0, stream>>>(x,      xb,     4096L * 2048);
    cvt_f32_bf16<<<2048, 256, 0, stream>>>(w_qkv,  wqkvb,  2048L * 6144);
    cvt_f32_bf16<<<1024, 256, 0, stream>>>(w_proj, wprojb, 2048L * 2048);

    gemm_bf16<true><<<dim3(6144 / 128, 4096 / 128), 256, 0, stream>>>(
        xb, wqkvb, qkvb, 4096, 6144, 2048);

    rope_reshape<<<(2 * 2048 * 16 * 64) / 256, 256, 0, stream>>>(
        qkvb, freqs, qbuf, kbuf, vbuf);

    attn_kernel<<<2 * 16 * (2048 / 64), 128, 0, stream>>>(qbuf, kbuf, vbuf, yb);

    gemm_bf16<false><<<dim3(2048 / 128, 4096 / 128), 256, 0, stream>>>(
        yb, wprojb, out, 4096, 2048, 2048);
}
